// LSTM_PP_74938589381015
// MI455X (gfx1250) — matrix operand order repaired and checked by reading, not compiled, not run on hardware
//
#include <hip/hip_runtime.h>
#include <hip/hip_bf16.h>

#define Bsz 256
#define Sd  512
#define Hd  512
#define Ed  128
#define TT  511   // S-1 timesteps

typedef __attribute__((ext_vector_type(16))) __bf16 v16bf;
typedef __attribute__((ext_vector_type(8)))  float  v8f;

union Frag {
    v16bf v;
    uint4 q[2];
};

// ---------------------------------------------------------------------------
// P0: embedding tables. type/state one-hot MLPs collapse to 4+3 vectors.
// ttab[4][E], stab[3][E]  (stab forced to 0 if use_mkt_state == 0)
// ---------------------------------------------------------------------------
__global__ __launch_bounds__(Ed) void embed_tables_kernel(
    const float* __restrict__ tw1, const float* __restrict__ tb1,
    const float* __restrict__ tw2, const float* __restrict__ tb2,
    const float* __restrict__ sw1, const float* __restrict__ sb1,
    const float* __restrict__ sw2, const float* __restrict__ sb2,
    const int*   __restrict__ use_mkt,
    float* __restrict__ ttab, float* __restrict__ stab)
{
    __shared__ float u[Ed];
    int j = threadIdx.x;
    for (int r = 0; r < 7; ++r) {
        bool isType = (r < 4);
        const float* w1 = isType ? tw1 : sw1;
        const float* b1 = isType ? tb1 : sb1;
        const float* w2 = isType ? tw2 : sw2;
        const float* b2 = isType ? tb2 : sb2;
        int row = isType ? r : (r - 4);
        u[j] = tanhf(w1[row * Ed + j] + b1[j]);
        __syncthreads();
        float s = b2[j];
        for (int i = 0; i < Ed; ++i) s += u[i] * w2[i * Ed + j];
        float v = tanhf(s);
        if (isType) ttab[r * Ed + j] = v;
        else        stab[(r - 4) * Ed + j] = (*use_mkt != 0) ? v : 0.0f;
        __syncthreads();
    }
}

// ---------------------------------------------------------------------------
// P1: xtab[c][g][n] = b1[g][n] + sum_i (ttab[e]+stab[s])[i] * gate_w1[g][i][n]
//     c = e*3 + s   (12 combos)
// ---------------------------------------------------------------------------
__global__ __launch_bounds__(Hd) void xtab_kernel(
    const float* __restrict__ ttab, const float* __restrict__ stab,
    const float* __restrict__ gw1,  const float* __restrict__ gb1,
    float* __restrict__ xtab)
{
    int cg = blockIdx.x;          // 0..47
    int c = cg >> 2, g = cg & 3;
    int e = c / 3, s = c % 3;
    int n = threadIdx.x;          // 0..511
    float acc = gb1[g * Hd + n];
    for (int i = 0; i < Ed; ++i) {
        float xv = ttab[e * Ed + i] + stab[s * Ed + i];
        acc += xv * gw1[((size_t)g * (Ed + Hd) + i) * Hd + n];
    }
    xtab[(c * 4 + g) * Hd + n] = acc;
}

// ---------------------------------------------------------------------------
// P2: transpose + convert weights to bf16, layout [g][n][k] so that the
//     WMMA B-fragment (K contiguous per column n) is two 16B loads.
//     W1h uses rows E..E+H of gate_w1 (the h-part of the concat).
// ---------------------------------------------------------------------------
__global__ __launch_bounds__(256) void convert_weights_kernel(
    const float* __restrict__ gw1, const float* __restrict__ gw2,
    __bf16* __restrict__ w1t, __bf16* __restrict__ w2t)
{
    int idx = blockIdx.x * blockDim.x + threadIdx.x;   // over 4*512*512
    if (idx >= 4 * Hd * Hd) return;
    int k = idx & 511;
    int n = (idx >> 9) & 511;
    int g = idx >> 18;
    w1t[idx] = (__bf16)gw1[((size_t)g * (Ed + Hd) + (Ed + k)) * Hd + n];
    w2t[idx] = (__bf16)gw2[((size_t)g * Hd + k) * Hd + n];
}

// ---------------------------------------------------------------------------
// P3: per-(b,s) combo index; zero c and h (recurrent state) every call.
// ---------------------------------------------------------------------------
__global__ __launch_bounds__(256) void init_state_kernel(
    const int* __restrict__ event, const int* __restrict__ mkt,
    unsigned char* __restrict__ comb, float* __restrict__ c,
    __bf16* __restrict__ hbf)
{
    int idx = blockIdx.x * blockDim.x + threadIdx.x;
    if (idx >= Bsz * Sd) return;                 // B*S == B*H == 131072
    comb[idx] = (unsigned char)(event[idx] * 3 + mkt[idx]);
    c[idx]   = 0.0f;
    hbf[idx] = (__bf16)0.0f;
}

// ---------------------------------------------------------------------------
// Layer 1: g_act[g][b][n] = tanh( h @ W1h[g] + xtab[comb[b][t]][g][n] )
// GEMM [256 x 512] x [512 x 2048]  (4 gates stacked along N)
// Wave tile: 64(M) x 32(N) = 4x2 WMMA tiles; 256 waves, 64 blocks of 4 waves.
// ---------------------------------------------------------------------------
__global__ __launch_bounds__(128) void lstm_layer1_kernel(
    const __bf16* __restrict__ hbf,          // [B][H] bf16
    const __bf16* __restrict__ w1t,          // [4][n][k] bf16
    const float*  __restrict__ xtab,         // [12][4][H]
    const unsigned char* __restrict__ comb,  // [B][S]
    __bf16* __restrict__ gact,               // [4][B][H] bf16
    int t)
{
    const int wave = (blockIdx.x * blockDim.x + threadIdx.x) >> 5;
    const int lane = threadIdx.x & 31;
    const int half = lane >> 4;
    const int lp   = lane & 15;
    const int m0 = (wave & 3) * 64;          // batch tile
    const int n0 = (wave >> 2) * 32;         // fused-gate column tile (0..2047)

    const v8f zacc = {0.f, 0.f, 0.f, 0.f, 0.f, 0.f, 0.f, 0.f};
    v8f acc[4][2];
    for (int mt = 0; mt < 4; ++mt)
        for (int nt = 0; nt < 2; ++nt) acc[mt][nt] = zacc;

    for (int kk = 0; kk < Hd; kk += 32) {
        Frag a[4], b[2];
        const int abase = kk + half * 8;     // A: K in {abase..+7} U {abase+16..+23}
        for (int mt = 0; mt < 4; ++mt) {
            const __bf16* p = hbf + (size_t)(m0 + mt * 16 + lp) * Hd + abase;
            a[mt].q[0] = *(const uint4*)p;
            a[mt].q[1] = *(const uint4*)(p + 16);
        }
        const int bbase = kk + half * 8;     // B: K in {bbase..+7} U {bbase+16..+23}
        for (int nt = 0; nt < 2; ++nt) {
            const int n = n0 + nt * 16 + lp;
            const int g = n >> 9, ni = n & 511;
            const __bf16* p = w1t + ((size_t)(g * Hd + ni)) * Hd + bbase;
            b[nt].q[0] = *(const uint4*)p;
            b[nt].q[1] = *(const uint4*)(p + 16);
        }
        for (int mt = 0; mt < 4; ++mt)
            for (int nt = 0; nt < 2; ++nt)
                acc[mt][nt] = __builtin_amdgcn_wmma_f32_16x16x32_bf16(
                    false, a[mt].v, false, b[nt].v, (short)0, acc[mt][nt],
                    false, false);
    }

    // epilogue: add xtab bias (per-row combo lookup), tanh, store bf16
    for (int nt = 0; nt < 2; ++nt) {
        const int n = n0 + nt * 16 + lp;
        const int g = n >> 9, ni = n & 511;
        for (int mt = 0; mt < 4; ++mt) {
            for (int i = 0; i < 8; ++i) {
                const int bb = m0 + mt * 16 + i + 8 * half;   // C/D row layout
                const int cmb = comb[bb * Sd + t];
                float v = acc[mt][nt][i] + xtab[(cmb * 4 + g) * Hd + ni];
                gact[((size_t)(g * Bsz + bb)) * Hd + ni] = (__bf16)tanhf(v);
            }
        }
    }
}

// ---------------------------------------------------------------------------
// Layer 2 + fused LSTM cell update.
// pre[g] = g_act[g] @ W2[g] + b2[g];  i,f,o = sigmoid, z = tanh;
// c' = f*c + i*z; h' = o*tanh(c'); write h' to out[b][t][:] and hbf (bf16).
// Wave tile: 32(M) x 16(N) per gate, all 4 gates => pointwise fuses in-regs.
// 256 waves, 64 blocks of 4 waves.
// ---------------------------------------------------------------------------
__global__ __launch_bounds__(128) void lstm_layer2_kernel(
    const __bf16* __restrict__ gact,  // [4][B][H] bf16
    const __bf16* __restrict__ w2t,   // [4][n][k] bf16
    const float*  __restrict__ b2,    // [4][H]
    float*  __restrict__ c,           // [B][H]
    __bf16* __restrict__ hbf,         // [B][H]
    float*  __restrict__ out,         // [B][511][H]
    int t)
{
    const int wave = (blockIdx.x * blockDim.x + threadIdx.x) >> 5;
    const int lane = threadIdx.x & 31;
    const int half = lane >> 4;
    const int lp   = lane & 15;
    const int m0 = (wave & 7) * 32;    // 8 M tiles
    const int n0 = (wave >> 3) * 16;   // 32 N tiles

    const v8f zacc = {0.f, 0.f, 0.f, 0.f, 0.f, 0.f, 0.f, 0.f};
    v8f acc[4][2];                     // [gate][mt]
    for (int g = 0; g < 4; ++g)
        for (int mt = 0; mt < 2; ++mt) acc[g][mt] = zacc;

    for (int kk = 0; kk < Hd; kk += 32) {
        const int abase = kk + half * 8;
        const int bbase = kk + half * 8;
        for (int g = 0; g < 4; ++g) {
            Frag bfr;
            const __bf16* pb = w2t + ((size_t)(g * Hd + n0 + lp)) * Hd + bbase;
            bfr.q[0] = *(const uint4*)pb;
            bfr.q[1] = *(const uint4*)(pb + 16);
            for (int mt = 0; mt < 2; ++mt) {
                Frag afr;
                const __bf16* pa =
                    gact + ((size_t)(g * Bsz + m0 + mt * 16 + lp)) * Hd + abase;
                afr.q[0] = *(const uint4*)pa;
                afr.q[1] = *(const uint4*)(pa + 16);
                acc[g][mt] = __builtin_amdgcn_wmma_f32_16x16x32_bf16(
                    false, afr.v, false, bfr.v, (short)0, acc[g][mt],
                    false, false);
            }
        }
    }

    const int n = n0 + lp;
    const float bb0 = b2[0 * Hd + n], bb1 = b2[1 * Hd + n];
    const float bb2v = b2[2 * Hd + n], bb3 = b2[3 * Hd + n];
    for (int mt = 0; mt < 2; ++mt) {
        for (int i = 0; i < 8; ++i) {
            const int bidx = m0 + mt * 16 + i + 8 * half;
            const float p0 = acc[0][mt][i] + bb0;
            const float p1 = acc[1][mt][i] + bb1;
            const float p2 = acc[2][mt][i] + bb2v;
            const float p3 = acc[3][mt][i] + bb3;
            const float ig = 1.0f / (1.0f + __expf(-p0));
            const float fg = 1.0f / (1.0f + __expf(-p1));
            const float og = 1.0f / (1.0f + __expf(-p2));
            const float z  = tanhf(p3);
            const size_t sidx = (size_t)bidx * Hd + n;
            const float cn = fg * c[sidx] + ig * z;
            c[sidx] = cn;
            const float h = og * tanhf(cn);
            out[((size_t)bidx * TT + t) * Hd + n] = h;
            hbf[sidx] = (__bf16)h;
        }
    }
}

// ---------------------------------------------------------------------------
extern "C" void kernel_launch(void* const* d_in, const int* in_sizes, int n_in,
                              void* d_out, int out_size, void* d_ws, size_t ws_size,
                              hipStream_t stream) {
    (void)in_sizes; (void)n_in; (void)out_size; (void)ws_size;
    const int*   event   = (const int*)d_in[0];
    const int*   mkt     = (const int*)d_in[1];
    const int*   use_mkt = (const int*)d_in[2];
    const float* tw1 = (const float*)d_in[3];
    const float* tb1 = (const float*)d_in[4];
    const float* tw2 = (const float*)d_in[5];
    const float* tb2 = (const float*)d_in[6];
    const float* sw1 = (const float*)d_in[7];
    const float* sb1 = (const float*)d_in[8];
    const float* sw2 = (const float*)d_in[9];
    const float* sb2 = (const float*)d_in[10];
    const float* gw1 = (const float*)d_in[11];
    const float* gb1 = (const float*)d_in[12];
    const float* gw2 = (const float*)d_in[13];
    const float* gb2 = (const float*)d_in[14];
    float* out = (float*)d_out;

    // workspace layout (all 256B aligned)
    char* ws = (char*)d_ws;
    size_t off = 0;
    auto alloc = [&](size_t bytes) {
        char* p = ws + off;
        off = (off + bytes + 255) & ~(size_t)255;
        return p;
    };
    __bf16* w1t  = (__bf16*)alloc((size_t)4 * Hd * Hd * 2);   // 2 MB
    __bf16* w2t  = (__bf16*)alloc((size_t)4 * Hd * Hd * 2);   // 2 MB
    float*  xtab = (float*) alloc((size_t)12 * 4 * Hd * 4);   // 96 KB
    float*  ttab = (float*) alloc((size_t)4 * Ed * 4);
    float*  stab = (float*) alloc((size_t)3 * Ed * 4);
    unsigned char* comb = (unsigned char*)alloc((size_t)Bsz * Sd); // 128 KB
    float*  cst  = (float*) alloc((size_t)Bsz * Hd * 4);      // 512 KB
    __bf16* hbf  = (__bf16*)alloc((size_t)Bsz * Hd * 2);      // 256 KB
    __bf16* gact = (__bf16*)alloc((size_t)4 * Bsz * Hd * 2);  // 1 MB

    // prep
    embed_tables_kernel<<<1, Ed, 0, stream>>>(tw1, tb1, tw2, tb2,
                                              sw1, sb1, sw2, sb2,
                                              use_mkt, ttab, stab);
    xtab_kernel<<<48, Hd, 0, stream>>>(ttab, stab, gw1, gb1, xtab);
    convert_weights_kernel<<<(4 * Hd * Hd + 255) / 256, 256, 0, stream>>>(
        gw1, gw2, w1t, w2t);
    init_state_kernel<<<(Bsz * Sd + 255) / 256, 256, 0, stream>>>(
        event, mkt, comb, cst, hbf);

    // sequential recurrence: 2 WMMA kernels per timestep (graph-captured)
    for (int t = 0; t < TT; ++t) {
        lstm_layer1_kernel<<<64, 128, 0, stream>>>(hbf, w1t, xtab, comb, gact, t);
        lstm_layer2_kernel<<<64, 128, 0, stream>>>(gact, w2t, gb2, cst, hbf, out, t);
    }
}
